// Attention_74406013435972
// MI455X (gfx1250) — matrix operand order repaired and checked by reading, not compiled, not run on hardware
//
#include <hip/hip_runtime.h>
#include <hip/hip_bf16.h>

typedef __attribute__((ext_vector_type(8)))  float  v8f;
typedef __attribute__((ext_vector_type(8)))  __bf16 v8bf;
typedef __attribute__((ext_vector_type(16))) __bf16 v16bf;
typedef int v4i __attribute__((vector_size(16)));

#define B_   2
#define S_   2048
#define D_   1024
#define H_   16
#define HD_  64
#define TD_  3072
#define PREFIX_ 256

#if __has_builtin(__builtin_amdgcn_global_load_async_to_lds_b128)
#define HAVE_ASYNC_LDS 1
#else
#define HAVE_ASYNC_LDS 0
#endif

// ---------------- fp32 -> bf16 conversion ----------------
__global__ void cvt_f32_bf16(const float* __restrict__ src,
                             __hip_bfloat16* __restrict__ dst, int n) {
    int i = blockIdx.x * blockDim.x + threadIdx.x;
    if (i < n) dst[i] = __float2bfloat16(src[i]);
}

// Build A fragment (16x32 bf16): lane&15 = M row; lane>>4 selects K halves.
__device__ inline v16bf load_a_frag(const __bf16* __restrict__ rowptr, int koff, int half) {
    v16bf a;
    v8bf a0 = *(const v8bf*)(rowptr + koff + half * 16);
    v8bf a1 = *(const v8bf*)(rowptr + koff + half * 16 + 8);
#pragma unroll
    for (int j = 0; j < 8; ++j) { a[j] = a0[j]; a[8 + j] = a1[j]; }
    return a;
}

#if HAVE_ASYNC_LDS
__device__ inline void async_cp16(const __bf16* g, __hip_bfloat16* l) {
    auto gp = reinterpret_cast<__attribute__((address_space(1))) v4i*>(
        (uintptr_t)g);
    auto lp = reinterpret_cast<__attribute__((address_space(3))) v4i*>(
        (uintptr_t)l);
    __builtin_amdgcn_global_load_async_to_lds_b128(gp, lp, 0, 0);
}
#endif

// ---------------- QKV GEMM: qkv = xb @ wqkv^T  (M=4096,N=3072,K=1024) ----------------
__global__ __launch_bounds__(256) void gemm_qkv(const __bf16* __restrict__ xb,
                                                const __bf16* __restrict__ wb,
                                                __hip_bfloat16* __restrict__ qb,
                                                __hip_bfloat16* __restrict__ kb,
                                                __hip_bfloat16* __restrict__ vtb) {
    const int lane = threadIdx.x & 31;
    const int wave = threadIdx.x >> 5;
    const int half = lane >> 4, l16 = lane & 15;
    const int m0 = blockIdx.x * 128 + wave * 16;
    const int n0 = blockIdx.y * 64;

    v8f acc[4] = {};
    const __bf16* arow = xb + (size_t)(m0 + l16) * D_;
    for (int k0 = 0; k0 < D_; k0 += 32) {
        v16bf a = load_a_frag(arow, k0, half);
#pragma unroll
        for (int f = 0; f < 4; ++f) {
            const __bf16* brow = wb + (size_t)(n0 + f * 16 + l16) * D_ + k0 + half * 16;
            v16bf b = *(const v16bf*)brow;
            acc[f] = __builtin_amdgcn_wmma_f32_16x16x32_bf16(
                false, a, false, b, (short)0, acc[f], false, false);
        }
    }
#pragma unroll
    for (int f = 0; f < 4; ++f) {
#pragma unroll
        for (int r = 0; r < 8; ++r) {
            int m = m0 + half * 8 + r;
            int n = n0 + f * 16 + l16;
            int b = m >> 11, s = m & (S_ - 1);
            __hip_bfloat16 val = __float2bfloat16(acc[f][r]);
            if (n < D_) {
                int h = n >> 6, d = n & 63;
                qb[(((size_t)(b * H_ + h)) * S_ + s) * HD_ + d] = val;
            } else if (n < 2 * D_) {
                int nn = n - D_; int h = nn >> 6, d = nn & 63;
                kb[(((size_t)(b * H_ + h)) * S_ + s) * HD_ + d] = val;
            } else {
                int nn = n - 2 * D_; int h = nn >> 6, d = nn & 63;
                vtb[(((size_t)(b * H_ + h)) * HD_ + d) * S_ + s] = val;
            }
        }
    }
}

// ---------------- Flash attention ----------------
// Block = 8 waves = 128 q rows of one (b,h). K/V tiles (64 cols) staged in LDS
// once per block (async double-buffered), shared by all 8 waves.
__global__ __launch_bounds__(256) void attn_kernel(const __bf16* __restrict__ qb,
                                                   const __bf16* __restrict__ kb,
                                                   const __bf16* __restrict__ vtb,
                                                   __hip_bfloat16* __restrict__ ctx) {
    __shared__ __hip_bfloat16 Kt[2][64][64];   // [buf][s_local][d]   8KB each
    __shared__ __hip_bfloat16 Vt[2][64][64];   // [buf][d][s_local]   8KB each
    __shared__ __hip_bfloat16 Pt[8][16][64];   // per-wave P tile     16KB
    const int tid  = threadIdx.x;
    const int lane = tid & 31;
    const int wave = tid >> 5;
    const int half = lane >> 4, l16 = lane & 15;
    const int h = blockIdx.y, b = blockIdx.z;
    const int q0 = blockIdx.x * 128 + wave * 16;
    const size_t bh = (size_t)b * H_ + h;
    const __bf16* qbase = qb + bh * S_ * HD_;
    const __bf16* kbase = kb + bh * S_ * HD_;
    const __bf16* vbase = vtb + bh * HD_ * S_;

    // staging assignment: thread copies 32B of K and 32B of V per k-block
    const int srow = tid >> 2;          // 0..63
    const int scol = (tid & 3) * 16;    // 0,16,32,48 (halfs)

    // uniform trip count for the whole block (extra cols masked below)
    int vis = blockIdx.x * 128 + 128;
    if (vis < PREFIX_) vis = PREFIX_;
    const int kend = vis;               // multiple of 64, <= S_

    // ---- stage k-block 0 into buffer 0 ----
    {
        const __bf16* gk = kbase + (size_t)srow * HD_ + scol;
        const __bf16* gv = vbase + (size_t)srow * S_ + scol;
#if HAVE_ASYNC_LDS
        async_cp16(gk,     &Kt[0][srow][scol]);
        async_cp16(gk + 8, &Kt[0][srow][scol + 8]);
        async_cp16(gv,     &Vt[0][srow][scol]);
        async_cp16(gv + 8, &Vt[0][srow][scol + 8]);
#else
        *(v16bf*)&Kt[0][srow][scol] = *(const v16bf*)gk;
        *(v16bf*)&Vt[0][srow][scol] = *(const v16bf*)gv;
#endif
    }

    // Q fragments, two d-chunks of 32
    v16bf aq[2];
#pragma unroll
    for (int c = 0; c < 2; ++c)
        aq[c] = load_a_frag(qbase + (size_t)(q0 + l16) * HD_, c * 32, half);

    float mi[8], li[8];
    v8f o[4] = {};
#pragma unroll
    for (int r = 0; r < 8; ++r) { mi[r] = -1e30f; li[r] = 0.f; }

    for (int kb0 = 0; kb0 < kend; kb0 += 64) {
        const int cur = (kb0 >> 6) & 1;
#if HAVE_ASYNC_LDS
#if __has_builtin(__builtin_amdgcn_s_wait_asynccnt)
        __builtin_amdgcn_s_wait_asynccnt(0);
#else
        asm volatile("s_wait_asynccnt 0" ::: "memory");
#endif
#endif
        __syncthreads();   // staging of buf 'cur' complete; buf '1-cur' free
        if (kb0 + 64 < kend) {
            const int nxt = 1 - cur;
            const __bf16* gk = kbase + (size_t)(kb0 + 64 + srow) * HD_ + scol;
            const __bf16* gv = vbase + (size_t)srow * S_ + (kb0 + 64) + scol;
#if HAVE_ASYNC_LDS
            async_cp16(gk,     &Kt[nxt][srow][scol]);
            async_cp16(gk + 8, &Kt[nxt][srow][scol + 8]);
            async_cp16(gv,     &Vt[nxt][srow][scol]);
            async_cp16(gv + 8, &Vt[nxt][srow][scol + 8]);
#else
            *(v16bf*)&Kt[nxt][srow][scol] = *(const v16bf*)gk;
            *(v16bf*)&Vt[nxt][srow][scol] = *(const v16bf*)gv;
#endif
        }

        // ---- scores S = Q K^T from LDS K tile ----
        v8f sc[4] = {};
#pragma unroll
        for (int c = 0; c < 2; ++c) {
#pragma unroll
            for (int f = 0; f < 4; ++f) {
                v16bf bk = *(const v16bf*)&Kt[cur][f * 16 + l16][c * 32 + half * 16];
                sc[f] = __builtin_amdgcn_wmma_f32_16x16x32_bf16(
                    false, aq[c], false, bk, (short)0, sc[f], false, false);
            }
        }
        // ---- mask + scale + row max ----
        float mnew[8];
#pragma unroll
        for (int r = 0; r < 8; ++r) {
            int qrow = q0 + half * 8 + r;
            float rowmax = -1e30f;
#pragma unroll
            for (int f = 0; f < 4; ++f) {
                int kcol = kb0 + f * 16 + l16;
                float s = sc[f][r] * 0.125f;          // 1/sqrt(64)
                bool ok = (kcol <= qrow) || (kcol < PREFIX_);
                s = ok ? s : -1e30f;
                sc[f][r] = s;
                rowmax = fmaxf(rowmax, s);
            }
#pragma unroll
            for (int msk = 1; msk < 16; msk <<= 1)
                rowmax = fmaxf(rowmax, __shfl_xor(rowmax, msk, 32));
            mnew[r] = fmaxf(mi[r], rowmax);
        }
        // ---- online softmax update + stage P (per-wave LDS) ----
#pragma unroll
        for (int r = 0; r < 8; ++r) {
            float alpha = __expf(mi[r] - mnew[r]);
            float rowsum = 0.f;
#pragma unroll
            for (int f = 0; f < 4; ++f) {
                float p = __expf(sc[f][r] - mnew[r]);
                sc[f][r] = p;
                rowsum += p;
            }
#pragma unroll
            for (int msk = 1; msk < 16; msk <<= 1)
                rowsum += __shfl_xor(rowsum, msk, 32);
            li[r] = li[r] * alpha + rowsum;
            mi[r] = mnew[r];
#pragma unroll
            for (int f = 0; f < 4; ++f) o[f][r] *= alpha;
#pragma unroll
            for (int f = 0; f < 4; ++f)
                Pt[wave][half * 8 + r][f * 16 + l16] = __float2bfloat16(sc[f][r]);
        }
        asm volatile("s_wait_dscnt 0" ::: "memory");   // wave-local LDS RAW fence
        // ---- O += P @ V from LDS V tile ----
#pragma unroll
        for (int c2 = 0; c2 < 2; ++c2) {
            v16bf ap = load_a_frag((const __bf16*)&Pt[wave][l16][0], c2 * 32, half);
#pragma unroll
            for (int f = 0; f < 4; ++f) {
                v16bf bv = *(const v16bf*)&Vt[cur][f * 16 + l16][c2 * 32 + half * 16];
                o[f] = __builtin_amdgcn_wmma_f32_16x16x32_bf16(
                    false, ap, false, bv, (short)0, o[f], false, false);
            }
        }
        asm volatile("" ::: "memory");
    }
    // ---- finalize ----
#pragma unroll
    for (int r = 0; r < 8; ++r) {
        int qrow = q0 + half * 8 + r;
        float inv = 1.f / li[r];
#pragma unroll
        for (int f = 0; f < 4; ++f)
            ctx[((size_t)b * S_ + qrow) * D_ + h * HD_ + f * 16 + l16] =
                __float2bfloat16(o[f][r] * inv);
    }
}

// ---------------- Output projection: out = ctx @ wproj^T + bias (fp32 out) ----------------
__global__ __launch_bounds__(256) void gemm_proj(const __bf16* __restrict__ ctx,
                                                 const __bf16* __restrict__ wb,
                                                 const float* __restrict__ bias,
                                                 float* __restrict__ out) {
    const int lane = threadIdx.x & 31;
    const int wave = threadIdx.x >> 5;
    const int half = lane >> 4, l16 = lane & 15;
    const int m0 = blockIdx.x * 128 + wave * 16;
    const int n0 = blockIdx.y * 64;

    v8f acc[4] = {};
    const __bf16* arow = ctx + (size_t)(m0 + l16) * D_;
    for (int k0 = 0; k0 < D_; k0 += 32) {
        v16bf a = load_a_frag(arow, k0, half);
#pragma unroll
        for (int f = 0; f < 4; ++f) {
            const __bf16* brow = wb + (size_t)(n0 + f * 16 + l16) * D_ + k0 + half * 16;
            v16bf b = *(const v16bf*)brow;
            acc[f] = __builtin_amdgcn_wmma_f32_16x16x32_bf16(
                false, a, false, b, (short)0, acc[f], false, false);
        }
    }
#pragma unroll
    for (int f = 0; f < 4; ++f) {
#pragma unroll
        for (int r = 0; r < 8; ++r) {
            int m = m0 + half * 8 + r;
            int n = n0 + f * 16 + l16;
            out[(size_t)m * D_ + n] = acc[f][r] + bias[n];
        }
    }
}

// ---------------- launch ----------------
extern "C" void kernel_launch(void* const* d_in, const int* in_sizes, int n_in,
                              void* d_out, int out_size, void* d_ws, size_t ws_size,
                              hipStream_t stream) {
    const float* x      = (const float*)d_in[0];
    const float* qkv_w  = (const float*)d_in[1];
    const float* proj_w = (const float*)d_in[2];
    const float* proj_b = (const float*)d_in[3];
    (void)in_sizes; (void)n_in; (void)out_size; (void)ws_size;

    char* ws = (char*)d_ws;
    __hip_bfloat16* xb    = (__hip_bfloat16*)(ws);                 //  8 MiB
    __hip_bfloat16* wqkv  = (__hip_bfloat16*)(ws + (8u  << 20));   //  6 MiB
    __hip_bfloat16* wproj = (__hip_bfloat16*)(ws + (14u << 20));   //  2 MiB
    __hip_bfloat16* qb    = (__hip_bfloat16*)(ws + (16u << 20));   //  8 MiB
    __hip_bfloat16* kb    = (__hip_bfloat16*)(ws + (24u << 20));   //  8 MiB
    __hip_bfloat16* vtb   = (__hip_bfloat16*)(ws + (32u << 20));   //  8 MiB
    __hip_bfloat16* ctx   = (__hip_bfloat16*)(ws + (40u << 20));   //  8 MiB

    int n_x = B_ * S_ * D_;
    int n_w1 = TD_ * D_;
    int n_w2 = D_ * D_;
    cvt_f32_bf16<<<(n_x  + 255) / 256, 256, 0, stream>>>(x, xb, n_x);
    cvt_f32_bf16<<<(n_w1 + 255) / 256, 256, 0, stream>>>(qkv_w, wqkv, n_w1);
    cvt_f32_bf16<<<(n_w2 + 255) / 256, 256, 0, stream>>>(proj_w, wproj, n_w2);

    gemm_qkv<<<dim3((B_ * S_) / 128, TD_ / 64), 256, 0, stream>>>(
        (const __bf16*)xb, (const __bf16*)wqkv, qb, kb, vtb);

    attn_kernel<<<dim3(S_ / 128, H_, B_), 256, 0, stream>>>(
        (const __bf16*)qb, (const __bf16*)kb, (const __bf16*)vtb, ctx);

    gemm_proj<<<dim3((B_ * S_) / 128, D_ / 64), 256, 0, stream>>>(
        (const __bf16*)ctx, (const __bf16*)wproj, proj_b, (float*)d_out);
}
